// SupportLockedPrimitiveRenderer_57964878627327
// MI455X (gfx1250) — matrix operand order repaired and checked by reading, not compiled, not run on hardware
//
#include <hip/hip_runtime.h>
#include <math.h>

// ---------------- problem constants ----------------
#define B_    8
#define H_    96
#define W_    96
#define HW    9216        // H_*W_
#define FC_   768
#define R_    12
#define P_    16
#define KS_   13
#define TOPK_ 24
#define NPIX  (B_*HW)     // 73728

typedef __attribute__((ext_vector_type(16))) __bf16 v16bf;
typedef __attribute__((ext_vector_type(8)))  float  v8f;

__device__ __forceinline__ float gelu_exact(float x) {
    return 0.5f * x * (1.0f + erff(x * 0.70710678118654752440f));
}

// =====================================================================
// Kernel 1: 1x1 conv FC->R (+GELU) as WMMA bf16 GEMM.
//   D[m,n] = sum_c w1[m,c] * feat[c,n]    (M=16 pad of R=12, N=16 px/wave, K=768)
// A fragment pre-swizzled into LDS in contiguous k order:
//   lane L, halfword i  ->  K = i + 16*(L>=16),            M = L&15
// B fragment in the same contiguous k order:
//   lane L, halfword i  ->  K = i + 16*(L>=16),            N = L&15
// Two independent accumulators break the WMMA C-chain; next K-block is
// prefetched (global_prefetch_b8) while the current one computes.
// =====================================================================
__global__ __launch_bounds__(256) void k_conv1_wmma(
        const float* __restrict__ feat, const float* __restrict__ w1,
        const float* __restrict__ b1, float* __restrict__ proj1) {
    __shared__ __attribute__((aligned(32))) __bf16 aswz[24][32][16]; // 24KB
    const int tid  = threadIdx.x;
    const int lane = tid & 31;
    const int wave = tid >> 5;

    // build swizzled A (w1) tiles: 24 K-blocks of 32
    for (int e = tid; e < 24 * 32 * 16; e += 256) {
        int i   = e & 15;
        int L   = (e >> 4) & 31;
        int blk = e >> 9;
        int m   = L & 15;
        int K   = blk * 32 + i + ((L >= 16) ? 16 : 0);
        float v = (m < R_) ? w1[m * FC_ + K] : 0.0f;
        aswz[blk][L][i] = (__bf16)v;
    }
    __syncthreads();

    const int n0  = (blockIdx.x * 8 + wave) * 16;  // 16 pixels per wave
    const int b   = n0 / HW;                       // tiles never cross batch
    const int sp  = n0 - b * HW;
    const float* fbase = feat + (size_t)b * FC_ * HW + sp;
    const int nlo   = lane & 15;
    const int choff = (lane >= 16) ? 16 : 0;

    v8f acc0 = {}, acc1 = {};
    for (int blk = 0; blk < 24; blk += 2) {
        // prefetch the next pair of K-blocks (64 channel rows, one per lane x2)
        if (blk + 2 < 24) {
            __builtin_prefetch(fbase + (size_t)((blk + 2) * 32 + lane) * HW, 0, 3);
            __builtin_prefetch(fbase + (size_t)((blk + 2) * 32 + 32 + lane) * HW, 0, 3);
        }
        // ---- even K-block -> acc0 ----
        {
            v16bf a = *(const v16bf*)(&aswz[blk][lane][0]);
            v16bf bb;
            const int cbase = blk * 32 + choff;
#pragma unroll
            for (int i = 0; i < 16; ++i)
                bb[i] = (__bf16)fbase[(size_t)(cbase + i) * HW + nlo];
            acc0 = __builtin_amdgcn_wmma_f32_16x16x32_bf16(
                    false, a, false, bb, (short)0, acc0, false, false);
        }
        // ---- odd K-block -> acc1 (independent of acc0 chain) ----
        {
            v16bf a = *(const v16bf*)(&aswz[blk + 1][lane][0]);
            v16bf bb;
            const int cbase = (blk + 1) * 32 + choff;
#pragma unroll
            for (int i = 0; i < 16; ++i)
                bb[i] = (__bf16)fbase[(size_t)(cbase + i) * HW + nlo];
            acc1 = __builtin_amdgcn_wmma_f32_16x16x32_bf16(
                    false, a, false, bb, (short)0, acc1, false, false);
        }
    }
    // epilogue: C layout VGPR j, lane L -> M = j + 8*(L>=16), N = L&15
#pragma unroll
    for (int j = 0; j < 8; ++j) {
        int m = j + ((lane >= 16) ? 8 : 0);
        if (m < R_) {
            float x = acc0[j] + acc1[j] + b1[m];
            proj1[((size_t)b * R_ + m) * HW + sp + nlo] = gelu_exact(x);
        }
    }
}

// =====================================================================
// Kernel 2: 3x3 conv R->R (+GELU) + abs-mean occupancy (VALU; tiny)
// =====================================================================
__global__ __launch_bounds__(256) void k_conv2(
        const float* __restrict__ proj1, const float* __restrict__ w2,
        const float* __restrict__ b2, float* __restrict__ proj,
        float* __restrict__ occ_raw) {
    __shared__ float sw[R_ * R_ * 9];
    for (int e = threadIdx.x; e < R_ * R_ * 9; e += 256) sw[e] = w2[e];
    __syncthreads();
    int n = blockIdx.x * 256 + threadIdx.x;
    if (n >= NPIX) return;
    int b = n / HW, sp = n % HW, y = sp / W_, x = sp % W_;
    const float* pb = proj1 + (size_t)b * R_ * HW;
    float asum = 0.0f;
    for (int ro = 0; ro < R_; ++ro) {
        float acc = b2[ro];
        for (int ri = 0; ri < R_; ++ri) {
            const float* pc = pb + (size_t)ri * HW;
            const float* wr = &sw[(ro * R_ + ri) * 9];
#pragma unroll
            for (int ky = 0; ky < 3; ++ky) {
                int yy = y + ky - 1;
                if (yy < 0 || yy >= H_) continue;
#pragma unroll
                for (int kx = 0; kx < 3; ++kx) {
                    int xx = x + kx - 1;
                    if (xx < 0 || xx >= W_) continue;
                    acc += wr[ky * 3 + kx] * pc[yy * W_ + xx];
                }
            }
        }
        float g = gelu_exact(acc);
        proj[((size_t)b * R_ + ro) * HW + sp] = g;
        asum += fabsf(g);
    }
    occ_raw[n] = asum / (float)R_;
}

// ------- per-batch min/max over HW elements (one WG per batch) -------
__global__ __launch_bounds__(256) void k_minmax(const float* __restrict__ x,
                                                float* __restrict__ mnmx) {
    __shared__ float smn[256], smx[256];
    int b = blockIdx.x, tid = threadIdx.x;
    const float* p = x + (size_t)b * HW;
    float mn = 1e30f, mx = -1e30f;
    for (int i = tid; i < HW; i += 256) {
        float v = p[i];
        mn = fminf(mn, v);
        mx = fmaxf(mx, v);
    }
    smn[tid] = mn; smx[tid] = mx;
    __syncthreads();
    for (int s = 128; s > 0; s >>= 1) {
        if (tid < s) {
            smn[tid] = fminf(smn[tid], smn[tid + s]);
            smx[tid] = fmaxf(smx[tid], smx[tid + s]);
        }
        __syncthreads();
    }
    if (tid == 0) { mnmx[b * 2] = smn[0]; mnmx[b * 2 + 1] = smx[0]; }
}

__device__ __forceinline__ float norm01(float v, float mn, float mx) {
    return (v - mn) / fmaxf(mx - mn, 1e-6f);
}

// ------- avgpool3 of |carrier| (count_include_pad: always /9) --------
__global__ __launch_bounds__(256) void k_cs(const float* __restrict__ carrier,
                                            float* __restrict__ cs_raw) {
    int n = blockIdx.x * 256 + threadIdx.x;
    if (n >= NPIX) return;
    int b = n / HW, sp = n % HW, y = sp / W_, x = sp % W_;
    const float* p = carrier + (size_t)b * HW;
    float s = 0.0f;
#pragma unroll
    for (int ky = -1; ky <= 1; ++ky) {
        int yy = y + ky; if (yy < 0 || yy >= H_) continue;
#pragma unroll
        for (int kx = -1; kx <= 1; ++kx) {
            int xx = x + kx; if (xx < 0 || xx >= W_) continue;
            s += fabsf(p[yy * W_ + xx]);
        }
    }
    cs_raw[n] = s / 9.0f;
}

// ------- evidence mix + threshold ------------------------------------
__global__ __launch_bounds__(256) void k_ev(
        const float* __restrict__ cs_raw, const float* __restrict__ dens,
        const float* __restrict__ mm_cs, const float* __restrict__ mm_d,
        float* __restrict__ m0) {
    int n = blockIdx.x * 256 + threadIdx.x;
    if (n >= NPIX) return;
    int b = n / HW;
    float cn = norm01(cs_raw[n], mm_cs[b * 2], mm_cs[b * 2 + 1]);
    float dn = norm01(dens[n],  mm_d[b * 2],  mm_d[b * 2 + 1]);
    float ev = 0.8f * cn + 0.2f * dn;
    m0[n] = (ev >= 0.28f) ? 1.0f : 0.0f;
}

// erosion = 1 - maxpool3(1-m)  (reduce_window pads -inf -> min over valid)
__global__ __launch_bounds__(256) void k_erode(const float* __restrict__ in,
                                               float* __restrict__ outp) {
    int n = blockIdx.x * 256 + threadIdx.x;
    if (n >= NPIX) return;
    int b = n / HW, sp = n % HW, y = sp / W_, x = sp % W_;
    const float* p = in + (size_t)b * HW;
    float mn = 1e30f;
#pragma unroll
    for (int ky = -1; ky <= 1; ++ky) {
        int yy = y + ky; if (yy < 0 || yy >= H_) continue;
#pragma unroll
        for (int kx = -1; kx <= 1; ++kx) {
            int xx = x + kx; if (xx < 0 || xx >= W_) continue;
            mn = fminf(mn, p[yy * W_ + xx]);
        }
    }
    outp[n] = mn;
}

// dilation = maxpool3 over valid, clamped to [0,1]
__global__ __launch_bounds__(256) void k_dilate(const float* __restrict__ in,
                                                float* __restrict__ outp) {
    int n = blockIdx.x * 256 + threadIdx.x;
    if (n >= NPIX) return;
    int b = n / HW, sp = n % HW, y = sp / W_, x = sp % W_;
    const float* p = in + (size_t)b * HW;
    float mx = -1e30f;
#pragma unroll
    for (int ky = -1; ky <= 1; ++ky) {
        int yy = y + ky; if (yy < 0 || yy >= H_) continue;
#pragma unroll
        for (int kx = -1; kx <= 1; ++kx) {
            int xx = x + kx; if (xx < 0 || xx >= W_) continue;
            mx = fmaxf(mx, p[yy * W_ + xx]);
        }
    }
    outp[n] = fminf(fmaxf(mx, 0.0f), 1.0f);
}

// scores = normalize(occ) * sup ; keep occn for extras
__global__ __launch_bounds__(256) void k_scores(
        const float* __restrict__ occ_raw, const float* __restrict__ mm_occ,
        const float* __restrict__ sup, float* __restrict__ occn,
        float* __restrict__ scores) {
    int n = blockIdx.x * 256 + threadIdx.x;
    if (n >= NPIX) return;
    int b = n / HW;
    float o = norm01(occ_raw[n], mm_occ[b * 2], mm_occ[b * 2 + 1]);
    occn[n] = o;
    scores[n] = o * sup[n];
}

// NMS peak mask applied to scores -> flat
__global__ __launch_bounds__(256) void k_peak(const float* __restrict__ scores,
                                              float* __restrict__ flat) {
    int n = blockIdx.x * 256 + threadIdx.x;
    if (n >= NPIX) return;
    int b = n / HW, sp = n % HW, y = sp / W_, x = sp % W_;
    const float* p = scores + (size_t)b * HW;
    float lm = -1e30f;
#pragma unroll
    for (int ky = -1; ky <= 1; ++ky) {
        int yy = y + ky; if (yy < 0 || yy >= H_) continue;
#pragma unroll
        for (int kx = -1; kx <= 1; ++kx) {
            int xx = x + kx; if (xx < 0 || xx >= W_) continue;
            lm = fmaxf(lm, p[yy * W_ + xx]);
        }
    }
    float s = p[sp];
    flat[n] = (s >= lm && s > 0.05f) ? s : 0.0f;
}

// top-24 per batch via iterative block-wide argmax in LDS
__global__ __launch_bounds__(256) void k_topk(
        const float* __restrict__ flat, float* __restrict__ vals,
        int* __restrict__ idxb, float* __restrict__ amap) {
    __shared__ float sv[HW];      // 36KB
    __shared__ float rv[256];
    __shared__ int   ri[256];
    int b = blockIdx.x, tid = threadIdx.x;
    const float* p = flat + (size_t)b * HW;
    for (int i = tid; i < HW; i += 256) {
        sv[i] = p[i];
        amap[(size_t)b * HW + i] = 0.0f;
    }
    __syncthreads();
    for (int k = 0; k < TOPK_; ++k) {
        float bm = -1e30f; int bi = 0;
        for (int i = tid; i < HW; i += 256) {
            float v = sv[i];
            if (v > bm) { bm = v; bi = i; }
        }
        rv[tid] = bm; ri[tid] = bi;
        __syncthreads();
        for (int s = 128; s > 0; s >>= 1) {
            if (tid < s) {
                bool take = (rv[tid + s] > rv[tid]) ||
                            (rv[tid + s] == rv[tid] && ri[tid + s] < ri[tid]);
                if (take) { rv[tid] = rv[tid + s]; ri[tid] = ri[tid + s]; }
            }
            __syncthreads();
        }
        if (tid == 0) {
            vals[b * TOPK_ + k] = rv[0];
            idxb[b * TOPK_ + k] = ri[0];
            amap[(size_t)b * HW + ri[0]] = rv[0];
            sv[ri[0]] = -1e30f;
        }
        __syncthreads();
    }
}

__global__ __launch_bounds__(256) void k_zero(float* __restrict__ p, int n) {
    int i = blockIdx.x * 256 + threadIdx.x;
    if (i < n) p[i] = 0.0f;
}

// one WG per anchor: prototype argmax + oriented-Gaussian stamp
__global__ __launch_bounds__(256) void k_stamp(
        const float* __restrict__ proj, const float* __restrict__ proto,
        const float* __restrict__ vals, const int* __restrict__ idxb,
        float* __restrict__ canvas) {
    __shared__ float sl[P_];
    __shared__ int spidx;
    int b = blockIdx.x / TOPK_, k = blockIdx.x % TOPK_;
    int tid = threadIdx.x;
    int id = idxb[b * TOPK_ + k];
    float val = vals[b * TOPK_ + k];
    if (tid < P_) {
        float acc = 0.0f;
        for (int r = 0; r < R_; ++r)
            acc += proj[((size_t)b * R_ + r) * HW + id] * proto[tid * R_ + r];
        sl[tid] = acc;
    }
    __syncthreads();
    if (tid == 0) {
        int bp = 0; float bv = sl[0];
        for (int p2 = 1; p2 < P_; ++p2)
            if (sl[p2] > bv) { bv = sl[p2]; bp = p2; }
        spidx = bp;
    }
    __syncthreads();
    int p = spidx;
    const float lenL[4] = {0.4f, 0.7f, 1.0f, 1.25f};
    const float widL[4] = {0.12f, 0.18f, 0.24f, 0.3f};
    float ori = (float)p * 3.14159265358979323846f / 16.0f;
    float ln = lenL[p & 3], wd = widL[(p >> 2) & 3];
    float c = cosf(ori), s = sinf(ori);
    int ys = id / W_, xs = id % W_;
    if (tid < KS_ * KS_) {
        int ky = tid / KS_, kx = tid % KS_;
        int yy = ys + ky - KS_ / 2, xx = xs + kx - KS_ / 2;
        if (yy >= 0 && yy < H_ && xx >= 0 && xx < W_) {
            float gx = -1.0f + (float)kx * (2.0f / 12.0f);
            float gy = -1.0f + (float)ky * (2.0f / 12.0f);
            float xr = c * gx + s * gy;
            float yr = -s * gx + c * gy;
            float foot = expf(-(xr / ln) * (xr / ln) - (yr / wd) * (yr / wd)) * val;
            for (int r = 0; r < R_; ++r)
                atomicAdd(&canvas[((size_t)b * R_ + r) * HW + yy * W_ + xx],
                          proto[p * R_ + r] * foot);
        }
    }
}

// =====================================================================
// Kernel: post-mix.  conv3 (16->12, VALU) + GELU, then conv4 (12->768)
// as WMMA bf16 GEMM (K padded 12->32), coalesced f32 stores of 226MB.
// hbuf staging is wave-private: only an s_wait_dscnt is needed, not a
// workgroup barrier.
// =====================================================================
__global__ __launch_bounds__(256) void k_postmix_wmma(
        const float* __restrict__ canvas, const float* __restrict__ occn,
        const float* __restrict__ sup, const float* __restrict__ amap,
        const float* __restrict__ dens, const float* __restrict__ mm_d,
        const float* __restrict__ w3, const float* __restrict__ b3,
        const float* __restrict__ w4, const float* __restrict__ b4,
        float* __restrict__ outp) {
    __shared__ __attribute__((aligned(32))) __bf16 a4[48][32][16];  // 48KB: w4 swizzled
    __shared__ __attribute__((aligned(32))) __bf16 hbuf[8][16][32]; // 8KB: per-wave B staging
    __shared__ float sw3[R_ * 16];
    __shared__ float sb3[R_];
    int tid = threadIdx.x, lane = tid & 31, wave = tid >> 5;

    for (int e = tid; e < 48 * 32 * 16; e += 256) {
        int i = e & 15;
        int L = (e >> 4) & 31;
        int t = e >> 9;
        int f = t * 16 + (L & 15);
        int K = i + ((L >= 16) ? 16 : 0);
        a4[t][L][i] = (__bf16)((K < R_) ? w4[f * R_ + K] : 0.0f);
    }
    for (int e = tid; e < R_ * 16; e += 256) sw3[e] = w3[e];
    if (tid < R_) sb3[tid] = b3[tid];
    __syncthreads();

    const int n0 = (blockIdx.x * 8 + wave) * 16;
    const int b  = n0 / HW;
    const int sp = n0 - b * HW;
    const float dmn = mm_d[b * 2], dmx = mm_d[b * 2 + 1];

    if (lane < 16) {                 // conv3 + GELU per pixel
        int n = n0 + lane;
        float supv = sup[n];
        float hin[16];
        for (int r = 0; r < R_; ++r)
            hin[r] = canvas[((size_t)b * R_ + r) * HW + sp + lane] * supv;
        hin[12] = occn[n];
        hin[13] = supv;
        hin[14] = amap[n];
        hin[15] = norm01(dens[n], dmn, dmx);
        for (int ro = 0; ro < R_; ++ro) {
            float acc = sb3[ro];
#pragma unroll
            for (int c = 0; c < 16; ++c) acc += sw3[ro * 16 + c] * hin[c];
            hbuf[wave][lane][ro] = (__bf16)gelu_exact(acc);
        }
        for (int ro = R_; ro < 32; ++ro) hbuf[wave][lane][ro] = (__bf16)0.0f;
    }
    // wave-private LDS staging: wait for this wave's DS stores only
    asm volatile("s_wait_dscnt 0x0" ::: "memory");

    // B fragment: lane L -> N = L&15, K = i + 16*(L>=16)  (K 12..31 are zero pad)
    v16bf bb = *(const v16bf*)(&hbuf[wave][lane & 15][(lane >= 16) ? 16 : 0]);

    for (int t = 0; t < 48; ++t) {
        v16bf a = *(const v16bf*)(&a4[t][lane][0]);
        v8f acc = {};
        acc = __builtin_amdgcn_wmma_f32_16x16x32_bf16(
                false, a, false, bb, (short)0, acc, false, false);
#pragma unroll
        for (int j = 0; j < 8; ++j) {
            int f = t * 16 + j + ((lane >= 16) ? 8 : 0);
            outp[((size_t)b * FC_ + f) * HW + sp + (lane & 15)] = acc[j] + b4[f];
        }
    }
}

// =====================================================================
// host launcher
// =====================================================================
extern "C" void kernel_launch(void* const* d_in, const int* in_sizes, int n_in,
                              void* d_out, int out_size, void* d_ws, size_t ws_size,
                              hipStream_t stream) {
    const float* features = (const float*)d_in[0];
    const float* carrier  = (const float*)d_in[1];
    const float* density  = (const float*)d_in[2];
    const float* w1    = (const float*)d_in[3];
    const float* b1    = (const float*)d_in[4];
    const float* w2    = (const float*)d_in[5];
    const float* b2    = (const float*)d_in[6];
    const float* w3    = (const float*)d_in[7];
    const float* b3    = (const float*)d_in[8];
    const float* w4    = (const float*)d_in[9];
    const float* b4    = (const float*)d_in[10];
    const float* proto = (const float*)d_in[11];
    float* outp = (float*)d_out;

    float* ws = (float*)d_ws;
    size_t off = 0;
    float* proj1   = ws + off; off += (size_t)B_ * R_ * HW;
    float* proj    = ws + off; off += (size_t)B_ * R_ * HW;
    float* canvas  = ws + off; off += (size_t)B_ * R_ * HW;
    float* occ_raw = ws + off; off += NPIX;
    float* occn    = ws + off; off += NPIX;
    float* cs_raw  = ws + off; off += NPIX;
    float* m0      = ws + off; off += NPIX;
    float* er      = ws + off; off += NPIX;
    float* m1      = ws + off; off += NPIX;
    float* sup     = ws + off; off += NPIX;
    float* scores  = ws + off; off += NPIX;
    float* flat    = ws + off; off += NPIX;
    float* amap    = ws + off; off += NPIX;
    float* mm_occ  = ws + off; off += 16;
    float* mm_cs   = ws + off; off += 16;
    float* mm_d    = ws + off; off += 16;
    float* vals    = ws + off; off += B_ * TOPK_;
    int*   idxb    = (int*)(ws + off); off += B_ * TOPK_;

    const int mapBlocks = (NPIX + 255) / 256;   // 288
    const int gemmBlocks = NPIX / 128;          // 576 (8 waves x 16 px each)

    k_conv1_wmma<<<gemmBlocks, 256, 0, stream>>>(features, w1, b1, proj1);
    k_conv2<<<mapBlocks, 256, 0, stream>>>(proj1, w2, b2, proj, occ_raw);

    k_minmax<<<B_, 256, 0, stream>>>(occ_raw, mm_occ);
    k_cs<<<mapBlocks, 256, 0, stream>>>(carrier, cs_raw);
    k_minmax<<<B_, 256, 0, stream>>>(cs_raw, mm_cs);
    k_minmax<<<B_, 256, 0, stream>>>(density, mm_d);

    k_ev<<<mapBlocks, 256, 0, stream>>>(cs_raw, density, mm_cs, mm_d, m0);
    k_erode<<<mapBlocks, 256, 0, stream>>>(m0, er);
    k_dilate<<<mapBlocks, 256, 0, stream>>>(er, m1);    // open: erode->dilate
    k_dilate<<<mapBlocks, 256, 0, stream>>>(m1, sup);   // dilation (DIL=3)

    k_scores<<<mapBlocks, 256, 0, stream>>>(occ_raw, mm_occ, sup, occn, scores);
    k_peak<<<mapBlocks, 256, 0, stream>>>(scores, flat);
    k_topk<<<B_, 256, 0, stream>>>(flat, vals, idxb, amap);

    const int canvasN = B_ * R_ * HW;
    k_zero<<<(canvasN + 255) / 256, 256, 0, stream>>>(canvas, canvasN);
    k_stamp<<<B_ * TOPK_, 256, 0, stream>>>(proj, proto, vals, idxb, canvas);

    k_postmix_wmma<<<gemmBlocks, 256, 0, stream>>>(
        canvas, occn, sup, amap, density, mm_d, w3, b3, w4, b4, outp);

    (void)in_sizes; (void)n_in; (void)out_size; (void)ws_size;
}
